// MixtureOfSoftMaxACF_4217657885003
// MI455X (gfx1250) — matrix operand order repaired and checked by reading, not compiled, not run on hardware
//
#include <hip/hip_runtime.h>

// MixtureOfSoftMax attention for MI455X (gfx1250, wave32, WMMA).
// B=8, d_k=512, N=N2=2048, N_MIX=2 (d=256), d_v=512.
// Outputs (concatenated): output [8,2048,512] f32, attn [8,2048,2048] f32.

typedef __attribute__((ext_vector_type(16))) __bf16 bf16x16;
typedef __attribute__((ext_vector_type(8)))  float  f32x8;
typedef __attribute__((ext_vector_type(4)))  float  f32x4;

union Frag {
    bf16x16 v;
    __bf16  h[16];
    uint4   q[2];
};

// ---------------------------------------------------------------------------
// Kernel 1: mixture gate pi = softmax(weight @ mean_N(qt)) -> d_ws[B*2]
// ---------------------------------------------------------------------------
__global__ void msm_pi_kernel(const float* __restrict__ qt,
                              const float* __restrict__ weight,
                              float* __restrict__ pi_ws) {
    constexpr int DK = 512, N = 2048;
    const int b = blockIdx.x;
    const int tid = threadIdx.x;

    __shared__ float barS[DK];
    __shared__ float red[256];

    const float* q = qt + (size_t)b * DK * N;
    for (int dd = tid; dd < DK; dd += 256) {
        const float4* row = (const float4*)(q + (size_t)dd * N);
        float s = 0.f;
        for (int n = 0; n < N / 4; ++n) {
            float4 x = row[n];
            s += (x.x + x.y) + (x.z + x.w);
        }
        barS[dd] = s * (1.0f / (float)N);
    }
    __syncthreads();

    float lg[2];
    for (int mi = 0; mi < 2; ++mi) {
        float p = weight[mi * DK + tid]       * barS[tid]
                + weight[mi * DK + tid + 256] * barS[tid + 256];
        red[tid] = p;
        __syncthreads();
        for (int s = 128; s > 0; s >>= 1) {
            if (tid < s) red[tid] += red[tid + s];
            __syncthreads();
        }
        lg[mi] = red[0];
        __syncthreads();
    }
    if (tid == 0) {
        float mx = fmaxf(lg[0], lg[1]);
        float e0 = __expf(lg[0] - mx), e1 = __expf(lg[1] - mx);
        float inv = 1.0f / (e0 + e1);
        pi_ws[b * 2 + 0] = e0 * inv;
        pi_ws[b * 2 + 1] = e1 * inv;
    }
}

// ---------------------------------------------------------------------------
// Kernel 2: one block = 16 query rows of one batch. Both mixtures' full
// 16x2048 logit tiles live in LDS (2*128KB) + bf16 Q tile (16KB) = 272KB.
// QK^T and attn@V use v_wmma_f32_16x16x32_bf16 (f32 accumulate).
// ---------------------------------------------------------------------------
__global__ void __launch_bounds__(256)
msm_attn_kernel(const float* __restrict__ qt,
                const float* __restrict__ kt,
                const float* __restrict__ vt,
                const float* __restrict__ pi_ws,
                float* __restrict__ out,        // [B,N,DV]
                float* __restrict__ attn_out) { // [B,N,N]
    constexpr int N = 2048, DK = 512, D = 256, DV = 512;
    const int b    = blockIdx.x >> 7;      // 128 row-blocks per batch
    const int row0 = (blockIdx.x & 127) * 16;
    const int tid  = threadIdx.x;
    const int wave = tid >> 5;
    const int lane = tid & 31;
    const int lh   = lane >> 4;            // lane half (0/1)
    const int ln   = lane & 15;

    extern __shared__ char smem[];
    __bf16* q_s = (__bf16*)smem;                              // [16][512] bf16
    float* l0 = (float*)(smem + 16 * DK * 2);                 // [16][2048] f32
    float* l1 = l0 + 16 * N;                                  // [16][2048] f32

    const float* qtb = qt + (size_t)b * DK * N;
    const float* ktb = kt + (size_t)b * DK * N;
    const float* vtb = vt + (size_t)b * DV * N;

    // ---- Phase A: Q tile -> LDS as bf16, layout q_s[r*512 + dd] ----
    for (int idx = tid; idx < 16 * DK; idx += 256) {
        int r = idx & 15, dd = idx >> 4;
        q_s[r * DK + dd] = (__bf16)qtb[(size_t)dd * N + row0 + r];
    }
    __syncthreads();

    const float invT = 0.04419417382415922f;   // 1/sqrt(512)

    // ---- Phase B: logits = (Q K^T)/temp for both mixtures -> LDS ----
    const __bf16* qrow = q_s + ln * DK;
    for (int mi = 0; mi < 2; ++mi) {
        Frag a[8];
#pragma unroll
        for (int kk = 0; kk < 8; ++kk) {
            // A 16x32 bf16 layout: two contiguous 8-elem runs per lane
            const int dd0 = mi * D + kk * 32 + (lh ? 8 : 0);
            a[kk].q[0] = *(const uint4*)(qrow + dd0);
            a[kk].q[1] = *(const uint4*)(qrow + dd0 + 16);
        }
        float* ldst = mi ? l1 : l0;
        for (int ct = wave; ct < 128; ct += 8) {
            const int col0 = ct * 16;
            __builtin_prefetch(ktb + (size_t)(mi * D) * N + col0 + 128 + ln, 0, 1);
            f32x8 acc = {};
#pragma unroll
            for (int kk = 0; kk < 8; ++kk) {
                Frag bb;
#pragma unroll
                for (int p = 0; p < 8; ++p) {
                    int k0 = (lh ? 8 : 0) + 2 * p + (p >= 4 ? 8 : 0);
                    const float* src =
                        ktb + (size_t)(mi * D + kk * 32 + k0) * N + col0 + ln;
                    bb.h[2 * p]     = (__bf16)src[0];
                    bb.h[2 * p + 1] = (__bf16)src[N];
                }
                acc = __builtin_amdgcn_wmma_f32_16x16x32_bf16(
                    false, a[kk].v, false, bb.v, (short)0, acc, false, false);
            }
#pragma unroll
            for (int r = 0; r < 8; ++r) {
                int mrow = r + (lh ? 8 : 0);
                ldst[mrow * N + col0 + ln] = acc[r] * invT;
            }
        }
    }
    __syncthreads();

    // ---- Phase C: per-row softmax (both mixtures), combine with pi ----
    const float pi0 = pi_ws[b * 2 + 0];
    const float pi1 = pi_ws[b * 2 + 1];
    for (int r = wave * 2; r < wave * 2 + 2; ++r) {
        float* row0p = l0 + r * N;
        float* row1p = l1 + r * N;
        float mx0 = -3.4e38f, mx1 = -3.4e38f;
        for (int c = lane * 4; c < N; c += 128) {
            float4 x0 = *(const float4*)(row0p + c);
            float4 x1 = *(const float4*)(row1p + c);
            mx0 = fmaxf(mx0, fmaxf(fmaxf(x0.x, x0.y), fmaxf(x0.z, x0.w)));
            mx1 = fmaxf(mx1, fmaxf(fmaxf(x1.x, x1.y), fmaxf(x1.z, x1.w)));
        }
#pragma unroll
        for (int o = 16; o > 0; o >>= 1) {
            mx0 = fmaxf(mx0, __shfl_xor(mx0, o, 32));
            mx1 = fmaxf(mx1, __shfl_xor(mx1, o, 32));
        }
        // exponentiate in place (so the combine pass is exp-free) and sum
        float s0 = 0.f, s1 = 0.f;
        for (int c = lane * 4; c < N; c += 128) {
            float4 x0 = *(const float4*)(row0p + c);
            float4 x1 = *(const float4*)(row1p + c);
            x0.x = __expf(x0.x - mx0); x0.y = __expf(x0.y - mx0);
            x0.z = __expf(x0.z - mx0); x0.w = __expf(x0.w - mx0);
            x1.x = __expf(x1.x - mx1); x1.y = __expf(x1.y - mx1);
            x1.z = __expf(x1.z - mx1); x1.w = __expf(x1.w - mx1);
            *(float4*)(row0p + c) = x0;
            *(float4*)(row1p + c) = x1;
            s0 += (x0.x + x0.y) + (x0.z + x0.w);
            s1 += (x1.x + x1.y) + (x1.z + x1.w);
        }
#pragma unroll
        for (int o = 16; o > 0; o >>= 1) {
            s0 += __shfl_xor(s0, o, 32);
            s1 += __shfl_xor(s1, o, 32);
        }
        const float w0 = pi0 / s0, w1 = pi1 / s1;
        float* arow = attn_out + ((size_t)b * N + row0 + r) * N;
        for (int c = lane * 4; c < N; c += 128) {
            float4 x0 = *(const float4*)(row0p + c);
            float4 x1 = *(const float4*)(row1p + c);
            f32x4 v;
            v.x = w0 * x0.x + w1 * x1.x;
            v.y = w0 * x0.y + w1 * x1.y;
            v.z = w0 * x0.z + w1 * x1.z;
            v.w = w0 * x0.w + w1 * x1.w;
            *(f32x4*)(row0p + c) = v;                  // combined attn for D
            __builtin_nontemporal_store(v, (f32x4*)(arow + c));  // NT stream
        }
    }
    __syncthreads();

    // ---- Phase D: output = attn @ V  (attn from LDS, V from global) ----
    f32x8 oacc[4] = {{}, {}, {}, {}};
    const float* lrow = l0 + ln * N;
    for (int kk2 = 0; kk2 < 64; ++kk2) {
        const int kbase = kk2 * 32;
        Frag a;
        {
            const int c0 = kbase + (lh ? 16 : 0);
            float4 x0 = *(const float4*)(lrow + c0);
            float4 x1 = *(const float4*)(lrow + c0 + 4);
            float4 x2 = *(const float4*)(lrow + c0 + 8);
            float4 x3 = *(const float4*)(lrow + c0 + 12);
            a.h[0] = (__bf16)x0.x;  a.h[1] = (__bf16)x0.y;
            a.h[2] = (__bf16)x0.z;  a.h[3] = (__bf16)x0.w;
            a.h[4] = (__bf16)x1.x;  a.h[5] = (__bf16)x1.y;
            a.h[6] = (__bf16)x1.z;  a.h[7] = (__bf16)x1.w;
            a.h[8] = (__bf16)x2.x;  a.h[9] = (__bf16)x2.y;
            a.h[10] = (__bf16)x2.z; a.h[11] = (__bf16)x2.w;
            a.h[12] = (__bf16)x3.x; a.h[13] = (__bf16)x3.y;
            a.h[14] = (__bf16)x3.z; a.h[15] = (__bf16)x3.w;
        }
#pragma unroll
        for (int j = 0; j < 4; ++j) {
            const int dv0 = (wave + j * 8) * 16;
            // B 32x16: per-lane K-run is 16 contiguous floats in vt
            const float* vrow =
                vtb + (size_t)(dv0 + ln) * N + kbase + (lh ? 16 : 0);
            float4 y0 = *(const float4*)(vrow);
            float4 y1 = *(const float4*)(vrow + 4);
            float4 y2 = *(const float4*)(vrow + 8);
            float4 y3 = *(const float4*)(vrow + 12);
            Frag bb;
            bb.h[0] = (__bf16)y0.x;  bb.h[1] = (__bf16)y0.y;
            bb.h[2] = (__bf16)y0.z;  bb.h[3] = (__bf16)y0.w;
            bb.h[4] = (__bf16)y1.x;  bb.h[5] = (__bf16)y1.y;
            bb.h[6] = (__bf16)y1.z;  bb.h[7] = (__bf16)y1.w;
            bb.h[8] = (__bf16)y2.x;  bb.h[9] = (__bf16)y2.y;
            bb.h[10] = (__bf16)y2.z; bb.h[11] = (__bf16)y2.w;
            bb.h[12] = (__bf16)y3.x; bb.h[13] = (__bf16)y3.y;
            bb.h[14] = (__bf16)y3.z; bb.h[15] = (__bf16)y3.w;
            oacc[j] = __builtin_amdgcn_wmma_f32_16x16x32_bf16(
                false, a.v, false, bb.v, (short)0, oacc[j], false, false);
        }
    }
#pragma unroll
    for (int j = 0; j < 4; ++j) {
        const int dv0 = (wave + j * 8) * 16;
#pragma unroll
        for (int r = 0; r < 8; ++r) {
            int mrow = r + (lh ? 8 : 0);
            __builtin_nontemporal_store(
                oacc[j][r], &out[((size_t)b * N + row0 + mrow) * DV + dv0 + ln]);
        }
    }
}

// ---------------------------------------------------------------------------
extern "C" void kernel_launch(void* const* d_in, const int* in_sizes, int n_in,
                              void* d_out, int out_size, void* d_ws, size_t ws_size,
                              hipStream_t stream) {
    (void)in_sizes; (void)n_in; (void)out_size; (void)ws_size;
    const float* qt     = (const float*)d_in[0];
    const float* kt     = (const float*)d_in[1];
    const float* vt     = (const float*)d_in[2];
    const float* weight = (const float*)d_in[3];

    float* out      = (float*)d_out;                        // [8,2048,512]
    float* attn_out = out + (size_t)8 * 2048 * 512;         // [8,2048,2048]
    float* pi_ws    = (float*)d_ws;                         // [16]

    msm_pi_kernel<<<8, 256, 0, stream>>>(qt, weight, pi_ws);

    const size_t lds = (size_t)16 * 512 * 2       // Q tile bf16
                     + (size_t)2 * 16 * 2048 * 4; // two logit tiles f32 = 272KB
    msm_attn_kernel<<<1024, 256, lds, stream>>>(qt, kt, vt, pi_ws, out, attn_out);
}
